// EdgeAttentionLayer_46617575031164
// MI455X (gfx1250) — matrix operand order repaired and checked by reading, not compiled, not run on hardware
//
#include <hip/hip_runtime.h>
#include <hip/hip_bf16.h>
#include <cstdint>

// ---------------- problem constants (match reference) ----------------
#define HIDDEN  256
#define NHEADS  8
#define HDIM    32     // HIDDEN / NHEADS
#define EDIM    64
#define SCALE   0.17677669529663687f   // HDIM^-0.5

typedef __attribute__((ext_vector_type(16))) __bf16 v16bf;
typedef __attribute__((ext_vector_type(8)))  __bf16 v8bf;
typedef __attribute__((ext_vector_type(8)))  float  v8f;

// ---------------- bf16 split helpers ----------------
__device__ __forceinline__ __bf16 bf_hi(float x) { return (__bf16)x; }
__device__ __forceinline__ __bf16 bf_lo(float x) {
    __bf16 h = (__bf16)x;
    return (__bf16)(x - (float)h);
}

// fp32-accurate product via three bf16 WMMAs: hi*hi + hi*lo + lo*hi
__device__ __forceinline__ v8f wmma3(const v16bf &ahi, const v16bf &alo,
                                     const v16bf &bhi, const v16bf &blo, v8f acc) {
    acc = __builtin_amdgcn_wmma_f32_16x16x32_bf16(false, ahi, false, bhi, (short)0, acc, false, false);
    acc = __builtin_amdgcn_wmma_f32_16x16x32_bf16(false, ahi, false, blo, (short)0, acc, false, false);
    acc = __builtin_amdgcn_wmma_f32_16x16x32_bf16(false, alo, false, bhi, (short)0, acc, false, false);
    return acc;
}

// A fragment from row-major bf16: lane l -> row = l&15, K-chunk select by half-wave:
//   lanes 0-15 : e=0..15 -> K=kb..kb+15  (kb=0)
//   lanes16-31 : same with kb=16
__device__ __forceinline__ v16bf load_a_frag(const __bf16* __restrict__ base, int k0) {
    v8bf c0 = *(const v8bf*)(base + k0);
    v8bf c1 = *(const v8bf*)(base + k0 + 8);
    return __builtin_shufflevector(c0, c1, 0,1,2,3,4,5,6,7,8,9,10,11,12,13,14,15);
}

// ---------------- weight pack kernel ----------------
// Converts a (256 x 256) row-major f32 weight into bf16 hi/lo stored in
// fragment-linear order: frag = (tile_base+nt)*8 + kblock; within a fragment,
// lane l (col = n0 + (l&15), kbase = 16*(l>>4)) owns 16 contiguous bf16 (K = kbase+e).
__global__ void pack_w_kernel(const float* __restrict__ W,
                              __bf16* __restrict__ hi, __bf16* __restrict__ lo,
                              int tile_base) {
    int i = blockIdx.x * blockDim.x + threadIdx.x;   // 16 tiles * 8 kblocks * 512
    if (i >= 16 * 8 * 512) return;
    const int e  = i & 15;
    const int l  = (i >> 4) & 31;
    const int kb = (i >> 9) & 7;
    const int nt = i >> 12;
    const int col  = nt * 16 + (l & 15);
    const int krow = kb * 32 + ((l & 16) ? 16 : 0) + e;
    const float v = W[(size_t)krow * HIDDEN + col];
    const size_t idx = ((size_t)(tile_base + nt) * 8 + kb) * 512 + (size_t)l * 16 + e;
    hi[idx] = bf_hi(v);
    lo[idx] = bf_lo(v);
}

// ---------------- x -> bf16 hi/lo (row-major) ----------------
__global__ void cvt_kernel(const float* __restrict__ x,
                           __bf16* __restrict__ hi, __bf16* __restrict__ lo, int total) {
    int i = blockIdx.x * blockDim.x + threadIdx.x;
    if (i >= total) return;
    const float v = x[i];
    hi[i] = bf_hi(v);
    lo[i] = bf_lo(v);
}

// ---------------- generic 16-row x (ntiles*16)-col GEMM body ----------------
// A: row-major bf16 hi/lo (lda = 256). B: packed fragment-order bf16 hi/lo.
// One wave: 16-row tile x 4 consecutive n-tiles, K = 256 in steps of 32.
__device__ __forceinline__ void gemm_wave(const __bf16* __restrict__ Ahi,
                                          const __bf16* __restrict__ Alo,
                                          const __bf16* __restrict__ Whi,
                                          const __bf16* __restrict__ Wlo,
                                          int arow, int ntg, int lane, v8f acc[4]) {
    const int akb = (lane & 16) ? 16 : 0;
    const __bf16* pa_hi = Ahi + (size_t)arow * HIDDEN + akb;
    const __bf16* pa_lo = Alo + (size_t)arow * HIDDEN + akb;
    for (int k0 = 0; k0 < HIDDEN; k0 += 32) {
        const v16bf ahi = load_a_frag(pa_hi, k0);
        const v16bf alo = load_a_frag(pa_lo, k0);
        #pragma unroll
        for (int t = 0; t < 4; ++t) {
            const size_t frag = ((size_t)(ntg + t) * 8 + (k0 >> 5)) * 512 + (size_t)lane * 16;
            const v16bf bhi = *(const v16bf*)(Whi + frag);
            const v16bf blo = *(const v16bf*)(Wlo + frag);
            acc[t] = wmma3(ahi, alo, bhi, blo, acc[t]);
        }
    }
}

// ---------------- kernel: fused QKV GEMM (N x 256) @ (256 x 768) ----------------
// grid: (N/16, 3), block: 128 (4 waves). Wave = 16-row tile x 4 n-tiles of 48.
__global__ void qkv_gemm_kernel(const __bf16* __restrict__ xhi, const __bf16* __restrict__ xlo,
                                const __bf16* __restrict__ whi, const __bf16* __restrict__ wlo,
                                float* __restrict__ Q, float* __restrict__ K,
                                float* __restrict__ V) {
    const int lane = threadIdx.x & 31;
    const int wave = threadIdx.x >> 5;
    const int m0   = blockIdx.x * 16;
    const int ntg  = (blockIdx.y * 4 + wave) * 4;   // first of 4 n-tiles (0..47)

    v8f acc[4] = {};
    gemm_wave(xhi, xlo, whi, wlo, m0 + (lane & 15), ntg, lane, acc);

    const int mb = m0 + 8 * (lane >> 4);   // D: vgpr g holds M = g + 8*(lane>=16)
    #pragma unroll
    for (int t = 0; t < 4; ++t) {
        const int nt = ntg + t;
        float* O = (nt < 16) ? Q : ((nt < 32) ? K : V);
        const int col = ((nt * 16) & (HIDDEN - 1)) + (lane & 15);
        #pragma unroll
        for (int g = 0; g < 8; ++g)
            O[(size_t)(mb + g) * HIDDEN + col] = acc[t][g];
    }
}

// ---------------- kernel: out = agg @ Wo + bo ----------------
// grid: (N/16, 1), block 128 (4 waves); wave handles 4 of the 16 n-tiles.
__global__ void out_gemm_kernel(const __bf16* __restrict__ ahi, const __bf16* __restrict__ alo,
                                const __bf16* __restrict__ whi, const __bf16* __restrict__ wlo,
                                const float* __restrict__ bo, float* __restrict__ out) {
    const int lane = threadIdx.x & 31;
    const int wave = threadIdx.x >> 5;
    const int m0   = blockIdx.x * 16;
    const int ntg  = wave * 4;

    v8f acc[4] = {};
    gemm_wave(ahi, alo, whi, wlo, m0 + (lane & 15), ntg, lane, acc);

    const int mb = m0 + 8 * (lane >> 4);
    #pragma unroll
    for (int t = 0; t < 4; ++t) {
        const int col = (ntg + t) * 16 + (lane & 15);
        const float b = bo[col];
        #pragma unroll
        for (int g = 0; g < 8; ++g)
            out[(size_t)(mb + g) * HIDDEN + col] = acc[t][g] + b;
    }
}

// ---------------- monotone float<->uint encoding for atomic max ----------------
__device__ __forceinline__ unsigned enc_f(float f) {
    unsigned u = __float_as_uint(f);
    return (u & 0x80000000u) ? ~u : (u | 0x80000000u);
}
__device__ __forceinline__ float dec_f(unsigned u) {
    return (u & 0x80000000u) ? __uint_as_float(u & 0x7fffffffu) : __uint_as_float(~u);
}

// ---------------- zero per-node accumulators ----------------
__global__ void init_kernel(unsigned* __restrict__ m_enc, float* __restrict__ den,
                            float* __restrict__ attn_sum, int n) {
    int i = blockIdx.x * blockDim.x + threadIdx.x;
    if (i < n) { m_enc[i] = 0u; den[i] = 0.f; attn_sum[i] = 0.f; }  // enc 0 == -inf identity
}

// ---------------- per-edge scores + segment max ----------------
__global__ void score_kernel(const int* __restrict__ ei, const float* __restrict__ ea,
                             const float* __restrict__ We,
                             const float* __restrict__ Q, const float* __restrict__ K,
                             float* __restrict__ score, unsigned* __restrict__ m_enc, int E) {
    int idx = blockIdx.x * blockDim.x + threadIdx.x;
    if (idx >= E * NHEADS) return;
    const int e = idx >> 3, h = idx & 7;
    const int src = ei[e];
    const int tgt = ei[E + e];
    const float4* q = (const float4*)(Q + (size_t)src * HIDDEN + h * HDIM);
    const float4* k = (const float4*)(K + (size_t)tgt * HIDDEN + h * HDIM);
    float dot = 0.f;
    #pragma unroll
    for (int i2 = 0; i2 < HDIM / 4; ++i2) {
        float4 a = q[i2], b = k[i2];
        dot += a.x * b.x + a.y * b.y + a.z * b.z + a.w * b.w;
    }
    float bias = 0.f;
    const float* row = ea + (size_t)e * EDIM;
    #pragma unroll 8
    for (int d = 0; d < EDIM; ++d) bias += row[d] * We[d * NHEADS + h];
    const float s = dot * SCALE + bias;
    score[idx] = s;
    atomicMax(m_enc + (size_t)tgt * NHEADS + h, enc_f(s));
}

// ---------------- softmax denominator ----------------
__global__ void den_kernel(const int* __restrict__ ei, const float* __restrict__ score,
                           const unsigned* __restrict__ m_enc, float* __restrict__ den, int E) {
    int idx = blockIdx.x * blockDim.x + threadIdx.x;
    if (idx >= E * NHEADS) return;
    const int e = idx >> 3, h = idx & 7;
    const int tgt = ei[E + e];
    const float m = dec_f(m_enc[(size_t)tgt * NHEADS + h]);
    atomicAdd(den + (size_t)tgt * NHEADS + h, expf(score[idx] - m));
}

// ---------------- per-node sum of attention weights ----------------
// Reference uses V[tgt] (constant within a segment), so agg[n] = V[n] * sum(attn).
__global__ void attnsum_kernel(const int* __restrict__ ei, const float* __restrict__ score,
                               const unsigned* __restrict__ m_enc, const float* __restrict__ den,
                               float* __restrict__ attn_sum, int E) {
    int idx = blockIdx.x * blockDim.x + threadIdx.x;
    if (idx >= E * NHEADS) return;
    const int e = idx >> 3, h = idx & 7;
    const int tgt = ei[E + e];
    const size_t nh = (size_t)tgt * NHEADS + h;
    const float ex = expf(score[idx] - dec_f(m_enc[nh]));
    atomicAdd(attn_sum + nh, ex / den[nh]);
}

// ---------------- agg = attn_sum[n,h] * V[n,h,:], emitted as bf16 hi/lo ----------------
__global__ void finalize_kernel(const float* __restrict__ V, const float* __restrict__ attn_sum,
                                __bf16* __restrict__ ahi, __bf16* __restrict__ alo, int total) {
    int i = blockIdx.x * blockDim.x + threadIdx.x;
    if (i >= total) return;
    const int c = i & (HIDDEN - 1);
    const int n = i >> 8;
    const float a = attn_sum[(size_t)n * NHEADS + (c >> 5)] * V[i];
    ahi[i] = bf_hi(a);
    alo[i] = bf_lo(a);
}

// ---------------- host-side launcher ----------------
extern "C" void kernel_launch(void* const* d_in, const int* in_sizes, int n_in,
                              void* d_out, int out_size, void* d_ws, size_t ws_size,
                              hipStream_t stream) {
    const float* x   = (const float*)d_in[0];
    const int*   ei  = (const int*)  d_in[1];   // (2,E): row0 = src, row1 = tgt
    const float* ea  = (const float*)d_in[2];
    const float* Wq  = (const float*)d_in[3];
    const float* Wk  = (const float*)d_in[4];
    const float* Wv  = (const float*)d_in[5];
    const float* We  = (const float*)d_in[6];
    const float* Wo  = (const float*)d_in[7];
    const float* bo  = (const float*)d_in[8];
    float* out = (float*)d_out;

    const int N = in_sizes[0] / HIDDEN;       // 50000 (divisible by 16)
    const int E = in_sizes[1] / 2;            // 320000
    const size_t NH = (size_t)N * HIDDEN;

    // ---- workspace layout ----
    float* ws = (float*)d_ws;
    float*    Q        = ws;                                      // N*256 f32
    float*    K        = Q + NH;                                  // N*256 f32
    float*    V        = K + NH;                                  // N*256 f32
    float*    score    = V + NH;                                  // E*8   f32
    unsigned* m_enc    = (unsigned*)(score + (size_t)E * NHEADS); // N*8
    float*    den      = (float*)(m_enc + (size_t)N * NHEADS);    // N*8
    float*    attn_sum = den + (size_t)N * NHEADS;                // N*8
    __bf16*   bfbase   = (__bf16*)(attn_sum + (size_t)N * NHEADS);
    __bf16*   xhi      = bfbase;                  // N*256 bf16
    __bf16*   xlo      = xhi + NH;                // N*256
    __bf16*   agghi    = xlo + NH;                // N*256
    __bf16*   agglo    = agghi + NH;              // N*256
    __bf16*   wqkv_hi  = agglo + NH;              // 48 tiles * 8 * 512 = 196608
    __bf16*   wqkv_lo  = wqkv_hi + 48 * 8 * 512;
    __bf16*   wo_hi    = wqkv_lo + 48 * 8 * 512;  // 16 tiles * 8 * 512 = 65536
    __bf16*   wo_lo    = wo_hi + 16 * 8 * 512;

    const int mtiles = N / 16;
    const int PACK_N = 16 * 8 * 512;              // elements per weight matrix

    // 0) pack weights (fragment order) + convert x to bf16 hi/lo
    pack_w_kernel<<<(PACK_N + 255) / 256, 256, 0, stream>>>(Wq, wqkv_hi, wqkv_lo, 0);
    pack_w_kernel<<<(PACK_N + 255) / 256, 256, 0, stream>>>(Wk, wqkv_hi, wqkv_lo, 16);
    pack_w_kernel<<<(PACK_N + 255) / 256, 256, 0, stream>>>(Wv, wqkv_hi, wqkv_lo, 32);
    pack_w_kernel<<<(PACK_N + 255) / 256, 256, 0, stream>>>(Wo, wo_hi, wo_lo, 0);
    cvt_kernel<<<((int)NH + 255) / 256, 256, 0, stream>>>(x, xhi, xlo, (int)NH);

    // 1) QKV projections (WMMA bf16x3)
    qkv_gemm_kernel<<<dim3(mtiles, 3), 128, 0, stream>>>(xhi, xlo, wqkv_hi, wqkv_lo, Q, K, V);

    // 2) zero accumulators
    {
        int n = N * NHEADS;
        init_kernel<<<(n + 255) / 256, 256, 0, stream>>>(m_enc, den, attn_sum, n);
    }

    // 3-5) scores + segment max, denominator, attention-weight sums
    {
        int n = E * NHEADS;
        score_kernel<<<(n + 255) / 256, 256, 0, stream>>>(ei, ea, We, Q, K, score, m_enc, E);
        den_kernel<<<(n + 255) / 256, 256, 0, stream>>>(ei, score, m_enc, den, E);
        attnsum_kernel<<<(n + 255) / 256, 256, 0, stream>>>(ei, score, m_enc, den, attn_sum, E);
    }

    // 6) agg = attn_sum * V  (directly as bf16 hi/lo for the next GEMM)
    finalize_kernel<<<((int)NH + 255) / 256, 256, 0, stream>>>(V, attn_sum, agghi, agglo, (int)NH);

    // 7) output projection + bias (WMMA bf16x3)
    out_gemm_kernel<<<dim3(mtiles, 1), 128, 0, stream>>>(agghi, agglo, wo_hi, wo_lo, bo, out);
}
